// AEALblock_2482491097124
// MI455X (gfx1250) — matrix operand order repaired and checked by reading, not compiled, not run on hardware
//
#include <hip/hip_runtime.h>
#include <hip/hip_bf16.h>
#include <stdint.h>

typedef unsigned short u16;
typedef __attribute__((ext_vector_type(16))) __bf16 v16bf;
typedef __attribute__((ext_vector_type(8)))  float  v8f;

union FragBF { v16bf v; unsigned u[8]; uint4 q[2]; };

__device__ __forceinline__ u16 f2bf(float x) {
  unsigned u = __float_as_uint(x);
  u += 0x7FFFu + ((u >> 16) & 1u);   // round-to-nearest-even
  return (u16)(u >> 16);
}

// ---------------------------------------------------------------------------
// bf16 WMMA GEMM, barrier-free: out = act(alpha*(A*B + bias) + res)
// A: [M,K] bf16 row-major (lda), M must be a multiple of 128 (true for all
//   calls here: M = 51200). B: pre-swizzled fragment-major bf16 (per 16-col
//   n-tile, per 32-deep k-step, per lane: 8 uints = exact WMMA B operand).
// BM=128 BN=64 BK=32; 256 thr = 8 waves; wave = one 16-row subtile x 4 n-tiles.
// A staging LDS is wave-private -> no __syncthreads anywhere. A global loads
// are software-pipelined one step ahead + deep-prefetched.
// ---------------------------------------------------------------------------
__global__ __launch_bounds__(256) void gemm_bf16(
    const u16* __restrict__ A, int lda,
    const unsigned* __restrict__ Bp, int M, int N, int K,
    const float* __restrict__ bias,
    const float* __restrict__ alpha_ptr,
    const float* __restrict__ res,
    float* __restrict__ outf, u16* __restrict__ outbf,
    int ldc, int act)
{
  __shared__ u16 As[8 * 512];              // 8 waves x (16 rows x 32 k)
  const int tid  = threadIdx.x;
  const int lane = tid & 31, wv = tid >> 5;
  const int ml = lane & 15, hh = lane >> 4;
  const int m0 = blockIdx.x * 128, n0 = blockIdx.y * 64;
  const int nks = K >> 5;
  const int nt0 = n0 >> 4;
  u16* Aseg = As + wv * 512;
  const int r = lane >> 1, part = lane & 1;
  const u16* Arow = A + (size_t)(m0 + wv * 16 + r) * lda + part * 16;
  u16* AsegSt = Aseg + r * 32 + part * 16;
  const u16* AsegLd = Aseg + ml * 32;
  v8f acc0 = {}, acc1 = {}, acc2 = {}, acc3 = {};

  // pipeline: preload A for k-step 0
  uint4 a0 = ((const uint4*)Arow)[0];
  uint4 a1 = ((const uint4*)Arow)[1];

  for (int ks = 0; ks < nks; ++ks) {
    // commit already-arrived A rows to wave-private LDS
    *(uint4*)(AsegSt)     = a0;
    *(uint4*)(AsegSt + 8) = a1;

    // issue next A tile load (uniform clamped index -> no branch) and a
    // deep prefetch of the A stream to cover HBM latency
    {
      int ksn = (ks + 1 < nks) ? (ks + 1) : ks;
      const uint4* gp = (const uint4*)(Arow + ksn * 32);
      a0 = gp[0]; a1 = gp[1];
      if (ks + 8 < nks) __builtin_prefetch(Arow + (ks + 8) * 32, 0, 3);
    }

    FragBF fa;                              // A 16x32 bf16 operand layout
    fa.q[0] = *(const uint4*)(AsegLd + 16 * hh);
    fa.q[1] = *(const uint4*)(AsegLd + 16 * hh + 8);

    FragBF fb0, fb1, fb2, fb3;              // pre-swizzled B straight from L2/WGP$
    {
      size_t b0o = ((size_t)(nt0 + 0) * nks + ks) * 256 + lane * 8;
      size_t b1o = ((size_t)(nt0 + 1) * nks + ks) * 256 + lane * 8;
      size_t b2o = ((size_t)(nt0 + 2) * nks + ks) * 256 + lane * 8;
      size_t b3o = ((size_t)(nt0 + 3) * nks + ks) * 256 + lane * 8;
      fb0.q[0] = *(const uint4*)(Bp + b0o); fb0.q[1] = *(const uint4*)(Bp + b0o + 4);
      fb1.q[0] = *(const uint4*)(Bp + b1o); fb1.q[1] = *(const uint4*)(Bp + b1o + 4);
      fb2.q[0] = *(const uint4*)(Bp + b2o); fb2.q[1] = *(const uint4*)(Bp + b2o + 4);
      fb3.q[0] = *(const uint4*)(Bp + b3o); fb3.q[1] = *(const uint4*)(Bp + b3o + 4);
    }
    acc0 = __builtin_amdgcn_wmma_f32_16x16x32_bf16(false, fa.v, false, fb0.v, (short)0, acc0, false, false);
    acc1 = __builtin_amdgcn_wmma_f32_16x16x32_bf16(false, fa.v, false, fb1.v, (short)0, acc1, false, false);
    acc2 = __builtin_amdgcn_wmma_f32_16x16x32_bf16(false, fa.v, false, fb2.v, (short)0, acc2, false, false);
    acc3 = __builtin_amdgcn_wmma_f32_16x16x32_bf16(false, fa.v, false, fb3.v, (short)0, acc3, false, false);
  }

  const float alpha = alpha_ptr ? alpha_ptr[0] : 1.0f;
  auto emit = [&](const v8f& acc, int t) {
    int gcol = n0 + t * 16 + ml;
    float bvf = bias ? bias[gcol] : 0.0f;
#pragma unroll
    for (int j = 0; j < 8; ++j) {
      int gm = m0 + wv * 16 + j + 8 * hh;
      float v = (acc[j] + bvf) * alpha;
      if (res) v += res[(size_t)gm * ldc + gcol];
      if (act == 1) v = fmaxf(v, 0.0f);
      else if (act == 2) v = (v > 0.0f) ? v : 0.1f * v;
      size_t o = (size_t)gm * ldc + gcol;
      if (outf)  outf[o]  = v;
      if (outbf) outbf[o] = f2bf(v);
    }
  };
  emit(acc0, 0); emit(acc1, 1); emit(acc2, 2); emit(acc3, 3);
}

// ---------------------------------------------------------------------------
// Weight pack: f32 weights -> bf16 fragment-major WMMA B layout.
// mode 0: w is (N,K) row-major (linear / 1x1 conv).
// mode 1: w is (64,64,3,3) OIHW, logical K index k = (ky*3+kx)*64 + ic.
// out uint idx = ((nt*(K/32)+ks)*32 + lane)*8 + j ; k = ks*32 + 16*(lane>>4) + 2*j
// ---------------------------------------------------------------------------
__global__ void pack_frag(const float* __restrict__ w, unsigned* __restrict__ out,
                          int N, int K, int mode)
{
  int idx = blockIdx.x * 256 + threadIdx.x;
  int total = (N * K) >> 1;
  if (idx >= total) return;
  int j    = idx & 7;
  int lane = (idx >> 3) & 31;
  int rest = idx >> 8;
  int nks = K >> 5;
  int ks = rest % nks, nt = rest / nks;
  int n = nt * 16 + (lane & 15);
  int k = ks * 32 + 16 * (lane >> 4) + 2 * j;
  float f0, f1;
  if (mode == 0) {
    f0 = w[(size_t)n * K + k];
    f1 = w[(size_t)n * K + k + 1];
  } else {
    int i0 = k & 63,       p0 = k >> 6;
    int i1 = (k + 1) & 63, p1 = (k + 1) >> 6;
    f0 = w[((size_t)n * 64 + i0) * 9 + p0];
    f1 = w[((size_t)n * 64 + i1) * 9 + p1];
  }
  out[idx] = (unsigned)f2bf(f0) | ((unsigned)f2bf(f1) << 16);
}

// ---------------------------------------------------------------------------
// NCHW f32 -> NHWC bf16 (+ optional f32 copy). One batch per blockIdx.z.
// ---------------------------------------------------------------------------
__global__ void chw2hwc_pack(const float* __restrict__ in, int C,
                             u16* __restrict__ outbf, int ldbf, int coloff,
                             float* __restrict__ outf, int ldf)
{
  __shared__ float tile[32][33];
  int b = blockIdx.z;
  int hw0 = blockIdx.x * 32;
  int c0  = blockIdx.y * 32;
  int tx = threadIdx.x, ty = threadIdx.y;
#pragma unroll
  for (int i = 0; i < 4; ++i) {
    int c = c0 + ty + i * 8;
    tile[ty + i * 8][tx] = in[((size_t)b * C + c) * 6400 + hw0 + tx];
  }
  __syncthreads();
#pragma unroll
  for (int i = 0; i < 4; ++i) {
    int hw = hw0 + ty + i * 8;
    size_t tok = (size_t)b * 6400 + hw;
    int c = c0 + tx;
    float v = tile[tx][ty + i * 8];
    outbf[tok * ldbf + coloff + c] = f2bf(v);
    if (outf) outf[tok * ldf + c] = v;
  }
}

// NHWC f32 [8,6400,256] -> NCHW f32 [8,256,6400]
__global__ void hwc2chw_f32(const float* __restrict__ in, float* __restrict__ out)
{
  __shared__ float tile[32][33];
  int b = blockIdx.z;
  int hw0 = blockIdx.x * 32;
  int c0  = blockIdx.y * 32;
  int tx = threadIdx.x, ty = threadIdx.y;
#pragma unroll
  for (int i = 0; i < 4; ++i) {
    int hw = hw0 + ty + i * 8;
    tile[ty + i * 8][tx] = in[((size_t)b * 6400 + hw) * 256 + c0 + tx];
  }
  __syncthreads();
#pragma unroll
  for (int i = 0; i < 4; ++i) {
    int c = c0 + ty + i * 8;
    out[((size_t)b * 256 + c) * 6400 + hw0 + tx] = tile[tx][ty + i * 8];
  }
}

// im2col 3x3 pad1 on NHWC bf16 [8,80,80,64] -> [51200, 576]
__global__ void im2col3(const u16* __restrict__ E, u16* __restrict__ X3)
{
  int idx = blockIdx.x * 256 + threadIdx.x;
  if (idx >= 51200 * 9) return;
  int t = idx / 9, p = idx % 9;
  int b = t / 6400, hw = t % 6400, h = hw / 80, w = hw % 80;
  int h2 = h + p / 3 - 1, w2 = w + p % 3 - 1;
  uint4* dst = (uint4*)(X3 + (size_t)t * 576 + p * 64);
  if (h2 >= 0 && h2 < 80 && w2 >= 0 && w2 < 80) {
    const uint4* s = (const uint4*)(E + ((size_t)b * 6400 + h2 * 80 + w2) * 64);
#pragma unroll
    for (int i = 0; i < 8; ++i) dst[i] = s[i];
  } else {
    uint4 z = make_uint4(0u, 0u, 0u, 0u);
#pragma unroll
    for (int i = 0; i < 8; ++i) dst[i] = z;
  }
}

// GroupNorm(groups=16, C=64) + LeakyReLU(0.1), NHWC f32 in -> bf16 out
__global__ __launch_bounds__(256) void gn_lrelu(const float* __restrict__ X,
    const float* __restrict__ s, const float* __restrict__ bb,
    u16* __restrict__ out)
{
  __shared__ float r1[256], r2[256];
  int b = blockIdx.x >> 4, g = blockIdx.x & 15;
  int c0 = g * 4;
  size_t base = (size_t)b * 6400 * 64;
  float sum = 0.f, ssq = 0.f;
  for (int t = threadIdx.x; t < 6400; t += 256) {
    size_t o = base + (size_t)t * 64 + c0;
#pragma unroll
    for (int c = 0; c < 4; ++c) { float v = X[o + c]; sum += v; ssq += v * v; }
  }
  r1[threadIdx.x] = sum; r2[threadIdx.x] = ssq;
  __syncthreads();
  for (int st = 128; st > 0; st >>= 1) {
    if (threadIdx.x < st) { r1[threadIdx.x] += r1[threadIdx.x + st]; r2[threadIdx.x] += r2[threadIdx.x + st]; }
    __syncthreads();
  }
  float mean = r1[0] * (1.0f / 25600.0f);
  float var  = r2[0] * (1.0f / 25600.0f) - mean * mean;
  float rstd = rsqrtf(var + 1e-5f);
  float sc[4], bc[4];
#pragma unroll
  for (int c = 0; c < 4; ++c) { sc[c] = s[c0 + c]; bc[c] = bb[c0 + c]; }
  for (int t = threadIdx.x; t < 6400; t += 256) {
    size_t o = base + (size_t)t * 64 + c0;
#pragma unroll
    for (int c = 0; c < 4; ++c) {
      float v = (X[o + c] - mean) * rstd * sc[c] + bc[c];
      v = (v > 0.f) ? v : 0.1f * v;
      out[o + c] = f2bf(v);
    }
  }
}

// LayerNorm over 256 channels; optional second input added; f32/bf16 outputs.
__global__ __launch_bounds__(256) void ln256(const float* __restrict__ A,
    const float* __restrict__ B2, const float* __restrict__ s, const float* __restrict__ bb,
    float* __restrict__ outf, u16* __restrict__ outbf)
{
  int wv = threadIdx.x >> 5, lane = threadIdx.x & 31;
  size_t t = (size_t)blockIdx.x * 8 + wv;
  const float* pa = A + t * 256 + lane * 8;
  float v[8];
  float sum = 0.f, ssq = 0.f;
#pragma unroll
  for (int i = 0; i < 8; ++i) {
    float x = pa[i];
    if (B2) x += B2[t * 256 + lane * 8 + i];
    v[i] = x; sum += x; ssq += x * x;
  }
#pragma unroll
  for (int m = 16; m > 0; m >>= 1) {
    sum += __shfl_xor(sum, m, 32);
    ssq += __shfl_xor(ssq, m, 32);
  }
  float mean = sum * (1.0f / 256.0f);
  float var  = ssq * (1.0f / 256.0f) - mean * mean;
  float rstd = rsqrtf(var + 1e-5f);
#pragma unroll
  for (int i = 0; i < 8; ++i) {
    int c = lane * 8 + i;
    float o = (v[i] - mean) * rstd * s[c] + bb[c];
    size_t idx = t * 256 + c;
    if (outf)  outf[idx]  = o;
    if (outbf) outbf[idx] = f2bf(o);
  }
}

// ---------------------------------------------------------------------------
// Windowed MHA core, one wave32 per (batch,head,qtile of 16 rows).
// S=400, dh=32, 4 heads. QKV bf16 [51200,384] (q|k|v; each 128 = 4 heads x 32).
// branch 0: row-band windows, branch 1: column-band windows (token remap).
// ---------------------------------------------------------------------------
__global__ __launch_bounds__(32) void attn_kernel(const u16* __restrict__ qkv,
                                                  u16* __restrict__ y, int branch)
{
  __shared__ float sc[16][400];
  __shared__ float rowsum[16];
  __shared__ u16 P[16 * 416];
  __shared__ u16 Qs[16 * 32];
  __shared__ u16 KV[32 * 32];
  int lane = threadIdx.x;
  int pair = blockIdx.x / 25, qt = blockIdx.x % 25;
  int head = pair & 3, batch = pair >> 2;
  int bb = batch >> 4, sub = batch & 15;
  auto tok = [&](int s2) -> int {
    if (branch == 0) { int h2 = s2 / 80, wc = s2 % 80; return (bb * 80 + sub * 5 + h2) * 80 + wc; }
    else             { int hr = s2 / 5,  w2 = s2 % 5;  return (bb * 80 + hr) * 80 + sub * 5 + w2; }
  };
  const int qoff = head * 32, koff = 128 + head * 32, voff = 256 + head * 32;
  const int ml = lane & 15, hh = lane >> 4;
  const int r2 = lane >> 1, part = lane & 1;   // row-pair staging ids

  { // stage Q tile: 2 lanes per row, 32B each (vectorized gather)
    int t = tok(qt * 16 + r2);
    const uint4* gp = (const uint4*)(qkv + (size_t)t * 384 + qoff + part * 16);
    uint4 a0 = gp[0], a1 = gp[1];
    *(uint4*)(Qs + r2 * 32 + part * 16)     = a0;
    *(uint4*)(Qs + r2 * 32 + part * 16 + 8) = a1;
  }
  FragBF fq;
  fq.q[0] = *(const uint4*)(Qs + ml * 32 + 16 * hh);
  fq.q[1] = *(const uint4*)(Qs + ml * 32 + 16 * hh + 8);

  for (int kt = 0; kt < 25; ++kt) {            // scores = Q * K^T
    {
      int t = tok(kt * 16 + r2);
      const uint4* gp = (const uint4*)(qkv + (size_t)t * 384 + koff + part * 16);
      uint4 a0 = gp[0], a1 = gp[1];
      *(uint4*)(KV + r2 * 32 + part * 16)     = a0;
      *(uint4*)(KV + r2 * 32 + part * 16 + 8) = a1;
    }
    FragBF fk;
    fk.q[0] = *(const uint4*)(KV + ml * 32 + 16 * hh);
    fk.q[1] = *(const uint4*)(KV + ml * 32 + 16 * hh + 8);
    v8f d = {};
    d = __builtin_amdgcn_wmma_f32_16x16x32_bf16(false, fq.v, false, fk.v, (short)0, d, false, false);
#pragma unroll
    for (int j = 0; j < 8; ++j) sc[j + 8 * hh][kt * 16 + ml] = d[j];
  }

  { // softmax: all 32 lanes; row = ml, each half-wave scans 200 cols
    const float scale = 0.17677669529663687f;  // 1/sqrt(32)
    int cbase = hh * 200;
    float mx = -1e30f;
    for (int k2 = 0; k2 < 200; ++k2) mx = fmaxf(mx, sc[ml][cbase + k2]);
    mx = fmaxf(mx, __shfl_xor(mx, 16, 32));
    float sum = 0.f;
    for (int k2 = 0; k2 < 200; ++k2) {
      float e = __expf((sc[ml][cbase + k2] - mx) * scale);
      sum += e;
      P[ml * 416 + cbase + k2] = f2bf(e);
    }
    sum += __shfl_xor(sum, 16, 32);
    if (hh == 0) {
      rowsum[ml] = sum;
      for (int k2 = 400; k2 < 416; ++k2) P[ml * 416 + k2] = 0;
    }
  }

  v8f o0 = {}, o1 = {};
  for (int ks = 0; ks < 13; ++ks) {            // out = P * V (K padded to 416)
#pragma unroll
    for (int p = 0; p < 2; ++p) {              // stage V transposed: KV[dh][seq]
      int rr = p * 16 + r2;
      int s2 = ks * 32 + rr;
      uint4 a0 = make_uint4(0u, 0u, 0u, 0u), a1 = a0;
      if (s2 < 400) {
        const uint4* gp = (const uint4*)(qkv + (size_t)tok(s2) * 384 + voff + part * 16);
        a0 = gp[0]; a1 = gp[1];
      }
      union { uint4 q; u16 e[8]; } ua0, ua1;
      ua0.q = a0; ua1.q = a1;
#pragma unroll
      for (int e = 0; e < 8; ++e) KV[(part * 16 + e) * 32 + rr]     = ua0.e[e];
#pragma unroll
      for (int e = 0; e < 8; ++e) KV[(part * 16 + 8 + e) * 32 + rr] = ua1.e[e];
    }
    FragBF fp;
    fp.q[0] = *(const uint4*)(P + ml * 416 + ks * 32 + 16 * hh);
    fp.q[1] = *(const uint4*)(P + ml * 416 + ks * 32 + 16 * hh + 8);
    FragBF fv0, fv1;
    fv0.q[0] = *(const uint4*)(KV + ml * 32 + 16 * hh);
    fv0.q[1] = *(const uint4*)(KV + ml * 32 + 16 * hh + 8);
    fv1.q[0] = *(const uint4*)(KV + (16 + ml) * 32 + 16 * hh);
    fv1.q[1] = *(const uint4*)(KV + (16 + ml) * 32 + 16 * hh + 8);
    o0 = __builtin_amdgcn_wmma_f32_16x16x32_bf16(false, fp.v, false, fv0.v, (short)0, o0, false, false);
    o1 = __builtin_amdgcn_wmma_f32_16x16x32_bf16(false, fp.v, false, fv1.v, (short)0, o1, false, false);
  }

#pragma unroll
  for (int j = 0; j < 8; ++j) {
    int m = j + 8 * hh;
    int t = tok(qt * 16 + m);
    float inv = 1.0f / rowsum[m];
    y[(size_t)t * 128 + head * 32 + ml]      = f2bf(o0[j] * inv);
    y[(size_t)t * 128 + head * 32 + 16 + ml] = f2bf(o1[j] * inv);
  }
}

// ---------------------------------------------------------------------------
extern "C" void kernel_launch(void* const* d_in, const int* in_sizes, int n_in,
                              void* d_out, int out_size, void* d_ws, size_t ws_size,
                              hipStream_t stream)
{
  const float* src   = (const float*)d_in[0];
  const float* feats = (const float*)d_in[1];
  const float* gamma = (const float*)d_in[2];
  const float* w0 = (const float*)d_in[3];  const float* b0 = (const float*)d_in[4];
  const float* w1 = (const float*)d_in[5];  const float* b1 = (const float*)d_in[6];
  const float* g1s= (const float*)d_in[7];  const float* g1b= (const float*)d_in[8];
  const float* w2 = (const float*)d_in[9];  const float* b2 = (const float*)d_in[10];
  const float* g2s= (const float*)d_in[11]; const float* g2b= (const float*)d_in[12];
  const float* w3 = (const float*)d_in[13]; const float* b3 = (const float*)d_in[14];
  const float* w4 = (const float*)d_in[15]; const float* b4 = (const float*)d_in[16];
  const float* wq1= (const float*)d_in[17]; const float* bq1= (const float*)d_in[18];
  const float* wo1= (const float*)d_in[19]; const float* bo1= (const float*)d_in[20];
  const float* wq2= (const float*)d_in[21]; const float* bq2= (const float*)d_in[22];
  const float* wo2= (const float*)d_in[23]; const float* bo2= (const float*)d_in[24];
  const float* l1w= (const float*)d_in[25]; const float* l1b= (const float*)d_in[26];
  const float* l2w= (const float*)d_in[27]; const float* l2b= (const float*)d_in[28];
  const float* n1s= (const float*)d_in[29]; const float* n1b= (const float*)d_in[30];
  const float* n2s= (const float*)d_in[31]; const float* n2b= (const float*)d_in[32];

  char* ws = (char*)d_ws;
  size_t o = 0;
  auto alloc = [&](size_t bytes) { size_t r = o; o += (bytes + 255) & ~(size_t)255; return r; };
  const size_t R0 = alloc(78643200);   // X0[51200,768]bf / X3[.,576]bf / QKV1+QKV2 / H1-lo
  const size_t R1 = alloc(26214400);   // T1 f32 / Y2 bf / H1-hi (contiguous with R0)
  const size_t R2 = alloc(13107200);   // T1 bf / Tt bf
  const size_t R3 = alloc(13107200);   // U f32[.,64] / Y1 bf
  const size_t R4 = alloc(6553600);    // E bf16[.,64]
  const size_t R5 = alloc(52428800);   // src NHWC f32
  const size_t R6 = alloc(52428800);   // S2 f32
  const size_t R7 = alloc(26214400);   // S2 bf / X bf
  const size_t R8 = alloc(52428800);   // SA f32
  const size_t R9 = alloc(52428800);   // X f32
  const size_t R10= alloc(52428800);   // FF f32 / LN2 out (in-place)
  const size_t W0p = alloc(98304*2),  W1p = alloc(8192*2),  W2p = alloc(36864*2);
  const size_t W3p = alloc(8192*2),   W4p = alloc(32768*2);
  const size_t Q1p = alloc(49152*2),  O1p = alloc(16384*2);
  const size_t Q2p = alloc(49152*2),  O2p = alloc(16384*2);
  const size_t L1p = alloc(262144*2), L2p = alloc(262144*2);

  u16*   X0     = (u16*)(ws + R0);
  u16*   X3     = (u16*)(ws + R0);
  u16*   QKV1   = (u16*)(ws + R0);
  u16*   QKV2   = (u16*)(ws + R0 + 39321600);
  u16*   H1     = (u16*)(ws + R0);
  float* T1f    = (float*)(ws + R1);
  u16*   Y2     = (u16*)(ws + R1);
  u16*   T1bf   = (u16*)(ws + R2);
  u16*   Ttbf   = (u16*)(ws + R2);
  float* Uf     = (float*)(ws + R3);
  u16*   Y1     = (u16*)(ws + R3);
  u16*   Ebf    = (u16*)(ws + R4);
  float* srcHWC = (float*)(ws + R5);
  float* S2f    = (float*)(ws + R6);
  u16*   S2bf   = (u16*)(ws + R7);
  u16*   Xbf    = (u16*)(ws + R7);
  float* SAf    = (float*)(ws + R8);
  float* Xf     = (float*)(ws + R9);
  float* FFf    = (float*)(ws + R10);

  // --- weight packs: f32 -> bf16 fragment-major WMMA-B layout ---
  auto packW = [&](const float* w, size_t off, int N, int K, int mode) {
    int n = (N * K) >> 1;
    pack_frag<<<(n + 255) / 256, 256, 0, stream>>>(w, (unsigned*)(ws + off), N, K, mode);
  };
  packW(w0,  W0p, 128, 768, 0);
  packW(w1,  W1p, 64,  128, 0);
  packW(w2,  W2p, 64,  576, 1);
  packW(w3,  W3p, 128, 64,  0);
  packW(w4,  W4p, 256, 128, 0);
  packW(wq1, Q1p, 384, 128, 0);
  packW(wo1, O1p, 128, 128, 0);
  packW(wq2, Q2p, 384, 128, 0);
  packW(wo2, O2p, 128, 128, 0);
  packW(l1w, L1p, 1024, 256, 0);
  packW(l2w, L2p, 256, 1024, 0);

  // --- NCHW -> NHWC: cat([src,feats]) bf16, plus src f32 copy ---
  dim3 tblk(32, 8);
  chw2hwc_pack<<<dim3(200, 8, 8),  tblk, 0, stream>>>(src,   256, X0, 768, 0,   srcHWC, 256);
  chw2hwc_pack<<<dim3(200, 16, 8), tblk, 0, stream>>>(feats, 512, X0, 768, 256, nullptr, 0);

  const int M = 51200;
  auto gemm = [&](const u16* A, int lda, size_t Bp, int N, int K,
                  const float* bias, const float* ap, const float* res,
                  float* outf, u16* outbf, int ldc, int act) {
    gemm_bf16<<<dim3(M / 128, N / 64), 256, 0, stream>>>(
        A, lda, (const unsigned*)(ws + Bp), M, N, K, bias, ap, res, outf, outbf, ldc, act);
  };

  // trans conv path
  gemm(X0,   768, W0p, 128, 768, b0, nullptr, nullptr, T1f, T1bf, 128, 0);
  gemm(T1bf, 128, W1p, 64,  128, b1, nullptr, nullptr, Uf, nullptr, 64, 0);
  gn_lrelu<<<128, 256, 0, stream>>>(Uf, g1s, g1b, Ebf);
  im2col3<<<(51200 * 9 + 255) / 256, 256, 0, stream>>>(Ebf, X3);
  gemm(X3,   576, W2p, 64,  576, b2, nullptr, nullptr, Uf, nullptr, 64, 0);
  gn_lrelu<<<128, 256, 0, stream>>>(Uf, g2s, g2b, Ebf);
  gemm(Ebf,  64,  W3p, 128, 64,  b3, nullptr, T1f,    nullptr, Ttbf, 128, 2);  // lrelu(x+res)
  gemm(Ttbf, 128, W4p, 256, 128, b4, gamma,   srcHWC, S2f, S2bf, 256, 0);       // gamma*t + src

  // windowed attention (both branches)
  gemm(S2bf,       256, Q1p, 384, 128, bq1, nullptr, nullptr, nullptr, QKV1, 384, 0);
  gemm(S2bf + 128, 256, Q2p, 384, 128, bq2, nullptr, nullptr, nullptr, QKV2, 384, 0);
  attn_kernel<<<12800, 32, 0, stream>>>(QKV1, Y1, 0);
  attn_kernel<<<12800, 32, 0, stream>>>(QKV2, Y2, 1);
  gemm(Y1, 128, O1p, 128, 128, bo1, nullptr, nullptr, SAf,       nullptr, 256, 0);
  gemm(Y2, 128, O2p, 128, 128, bo2, nullptr, nullptr, SAf + 128, nullptr, 256, 0);

  // LN1(x + sa), FFN, LN2, final transpose
  ln256<<<6400, 256, 0, stream>>>(S2f, SAf, n1s, n1b, Xf, Xbf);
  gemm(Xbf, 256,  L1p, 1024, 256,  l1b, nullptr, nullptr, nullptr, H1, 1024, 1); // relu
  gemm(H1,  1024, L2p, 256,  1024, l2b, nullptr, Xf,      FFf, nullptr, 256, 0); // +x
  ln256<<<6400, 256, 0, stream>>>(FFf, nullptr, n2s, n2b, FFf, nullptr);
  hwc2chw_f32<<<dim3(200, 8, 8), tblk, 0, stream>>>(FFf, (float*)d_out);

  (void)in_sizes; (void)n_in; (void)out_size; (void)ws_size;
}
